// IntentionDecoder_14903536517872
// MI455X (gfx1250) — compile-verified, matrix operand order checked by reading, not run on hardware
//
#include <hip/hip_runtime.h>
#include <hip/hip_bf16.h>

// ---------------------------------------------------------------------------
// Problem constants
// ---------------------------------------------------------------------------
#define BATCH  4096
#define DIMD   256
#define HID    512
#define GATES  2048          // 4*HID
#define TSTEPS 45

typedef __attribute__((ext_vector_type(16))) __bf16 v16bf;
typedef __attribute__((ext_vector_type(8)))  __bf16 v8bf;
typedef __attribute__((ext_vector_type(8)))  float  v8f;

#define WMMA_BF16(A, B, C) \
    __builtin_amdgcn_wmma_f32_16x16x32_bf16(false, (A), false, (B), (short)0, (C), false, false)

// Opaque SGPR barrier: forces the compiler to treat a uniform pointer as
// redefined here, blocking LICM from hoisting (and then spilling) the loads
// derived from it out of the time loop.
#define OPAQUE_PTR(p) asm volatile("" : "+s"(p))

// Explicit global-address-space views (ptr->int->ptr keeps clang happy and is
// exact on amdgcn: global addresses == flat addresses). This restores
// global_load_* codegen after OPAQUE_PTR erased pointer provenance.
typedef const __attribute__((address_space(1))) v8bf*   gp_v8bf;
typedef const __attribute__((address_space(1))) float*  gp_f32;

__device__ __forceinline__ float ldg_f(const float* p) {
    return *(gp_f32)(unsigned long long)p;
}

// Workspace layout (bytes)
static constexpr size_t SZ_WT  = (size_t)GATES * HID * 2;        // 2 MB per transposed bf16 weight
static constexpr size_t OFF_WT0 = 0;                             // W_hh0^T
static constexpr size_t OFF_WT1 = OFF_WT0 + SZ_WT;               // W_ih1^T
static constexpr size_t OFF_WT2 = OFF_WT1 + SZ_WT;               // W_hh1^T
static constexpr size_t OFF_GB  = OFF_WT2 + SZ_WT;               // gates0_base (B x 2048 f32)
static constexpr size_t SZ_GB   = (size_t)BATCH * GATES * 4;
static constexpr size_t SZ_ST   = (size_t)BATCH * HID * 4;       // one f32 state
static constexpr size_t OFF_H0  = OFF_GB + SZ_GB;
static constexpr size_t OFF_C0  = OFF_H0 + SZ_ST;
static constexpr size_t OFF_H1  = OFF_C0 + SZ_ST;
static constexpr size_t OFF_C1  = OFF_H1 + SZ_ST;

__device__ __forceinline__ float fsig(float x)  { return 1.0f / (1.0f + __expf(-x)); }
__device__ __forceinline__ float ftanh(float x) { return 1.0f - 2.0f / (__expf(2.0f * x) + 1.0f); }

// LDS fragment load: two contiguous 16B groups -> ds_load_b128 x2
__device__ __forceinline__ v16bf load_frag_lds(const __bf16* p) {
    union { v16bf v; v8bf h[2]; } u;
    u.h[0] = *(const v8bf*)(p);
    u.h[1] = *(const v8bf*)(p + 16);
    return u.v;
}

// Global fragment load (addrspace(1)): global_load_b128 x2, LOADcnt only
__device__ __forceinline__ v16bf load_frag_g(const __bf16* p) {
    union { v16bf v; v8bf h[2]; } u;
    gp_v8bf q = (gp_v8bf)(unsigned long long)p;
    u.h[0] = q[0];
    u.h[1] = q[2];          // +2 x 16B = +16 bf16 elements
    return u.v;
}

// ---------------------------------------------------------------------------
// Prep kernel 1: transpose + convert the three recurrent 512x2048 weights into
// bf16 [N][K] layout so WMMA B-fragments are contiguous 16B loads.
// ---------------------------------------------------------------------------
__global__ __launch_bounds__(256) void k_prep_weights(const float* __restrict__ Whh0,
                                                      const float* __restrict__ Wih1,
                                                      const float* __restrict__ Whh1,
                                                      __bf16* __restrict__ wt) {
    const float* src = (blockIdx.y == 0) ? Whh0 : (blockIdx.y == 1) ? Wih1 : Whh1;
    __bf16* dst = wt + (size_t)blockIdx.y * GATES * HID;
    int idx = blockIdx.x * 256 + threadIdx.x;     // n*512 + k
    int n = idx >> 9;
    int k = idx & 511;
    dst[idx] = (__bf16)src[(size_t)k * GATES + n];
}

// ---------------------------------------------------------------------------
// Prep kernel 2: h_init / c_init  =  encoder_feat @ {Wh,Wc}_init + bias
// ---------------------------------------------------------------------------
__global__ __launch_bounds__(256) void k_init_state(const float* __restrict__ enc,
                                                    const float* __restrict__ Wh, const float* __restrict__ bh,
                                                    const float* __restrict__ Wc, const float* __restrict__ bc,
                                                    float* __restrict__ H0, float* __restrict__ C0,
                                                    float* __restrict__ H1, float* __restrict__ C1) {
    int idx = blockIdx.x * 256 + threadIdx.x;     // b*1024 + col
    int b = idx >> 10;
    int col = idx & 1023;
    float sh = bh[col], sc = bc[col];
    const float* e = enc + (size_t)b * DIMD;
    #pragma unroll 4
    for (int d = 0; d < DIMD; ++d) {
        float ev = e[d];
        sh += ev * Wh[(size_t)d * 1024 + col];
        sc += ev * Wc[(size_t)d * 1024 + col];
    }
    int l = col >> 9, j = col & 511;
    float* Hp = l ? H1 : H0;
    float* Cp = l ? C1 : C0;
    Hp[(size_t)b * HID + j] = sh;
    Cp[(size_t)b * HID + j] = sc;
}

// ---------------------------------------------------------------------------
// Prep kernel 3: gates0_base = context @ W_ih0[1:,:] + b_ih0 + b_hh0
// (context is time-invariant, so this is hoisted out of the T loop)
// ---------------------------------------------------------------------------
__global__ __launch_bounds__(256) void k_init_gbase(const float* __restrict__ ctx,
                                                    const float* __restrict__ Wih0,
                                                    const float* __restrict__ b0,
                                                    const float* __restrict__ b1,
                                                    float* __restrict__ gbase) {
    int idx = blockIdx.x * 256 + threadIdx.x;     // b*2048 + n
    int b = idx >> 11;
    int n = idx & 2047;
    float s = b0[n] + b1[n];
    const float* c = ctx + (size_t)b * DIMD;
    #pragma unroll 4
    for (int d = 0; d < DIMD; ++d)
        s += c[d] * Wih0[(size_t)(1 + d) * GATES + n];
    gbase[idx] = s;
}

// ---------------------------------------------------------------------------
// Main persistent kernel: one workgroup (512 threads = 16 wave32) owns 16
// batch rows for all 45 steps. Wave w owns hidden columns [w*32, w*32+32):
// 2 subtiles x 4 gates = 8 WMMA accumulator tiles (64 VGPRs) per wave.
// h0/h1 live in LDS (bf16 A-operands); c0/c1 live in registers in the exact
// WMMA C/D lane layout. Weight loads are re-issued every step (L2-resident,
// global_load_b128) behind an opaque pointer barrier so LICM cannot hoist +
// spill them across the time loop.
// ---------------------------------------------------------------------------
__global__ __launch_bounds__(512, 1) void lstm_persistent(
    const float* __restrict__ gbase_in,
    const float* __restrict__ w0row_in,       // W_ih0 row 0 (intent column), 2048 f32
    const __bf16* __restrict__ WT_hh0,
    const __bf16* __restrict__ WT_ih1,
    const __bf16* __restrict__ WT_hh1,
    const float* __restrict__ b_ih1_in, const float* __restrict__ b_hh1_in,
    const float* __restrict__ H0, const float* __restrict__ C0,
    const float* __restrict__ H1, const float* __restrict__ C1,
    const float* __restrict__ init_intent,
    const float* __restrict__ Wo1_in, const float* __restrict__ bo1_in,
    const float* __restrict__ Wo2_in, const float* __restrict__ bo2_in,
    const float* __restrict__ Wg1, const float* __restrict__ bg1,
    const float* __restrict__ Wg2, const float* __restrict__ bg2,
    float* __restrict__ out_steps, float* __restrict__ out_global) {

    __shared__ __align__(16) __bf16 h0s[16 * HID];
    __shared__ __align__(16) __bf16 h1s[16 * HID];
    __shared__ float intent_s[16];

    const int tid   = threadIdx.x;
    const int wave  = tid >> 5;       // 0..15
    const int lane  = tid & 31;
    const int lhalf = lane >> 4;      // selects K(+8) half / M(+8) half
    const int ln    = lane & 15;      // N (B/C tiles) or M (A tiles) within tile
    const int ko8   = lhalf * 8;
    const int row0  = blockIdx.x * 16;
    const int hbase = wave * 32;      // 32-column hidden slice per wave

    // --- load initial hidden states to LDS (bf16)
    for (int idx = tid; idx < 16 * HID; idx += 512) {
        int m = idx >> 9, j = idx & 511;
        h0s[idx] = (__bf16)H0[(size_t)(row0 + m) * HID + j];
        h1s[idx] = (__bf16)H1[(size_t)(row0 + m) * HID + j];
    }
    if (tid < 16) intent_s[tid] = init_intent[row0 + tid];

    // --- cell states in registers, WMMA C/D layout: vgpr r, lane -> (m = r+8*lhalf, hcol)
    v8f c0r[2], c1r[2];
    #pragma unroll
    for (int st = 0; st < 2; ++st) {
        int hcol = hbase + st * 16 + ln;
        #pragma unroll
        for (int r = 0; r < 8; ++r) {
            int m = r + 8 * lhalf;
            c0r[st][r] = C0[(size_t)(row0 + m) * HID + hcol];
            c1r[st][r] = C1[(size_t)(row0 + m) * HID + hcol];
        }
    }
    __syncthreads();

    for (int t = 0; t < TSTEPS; ++t) {
        // --- opaque copies of all time-invariant pointers: kills cross-step LICM
        const __bf16* wt0 = WT_hh0;
        const __bf16* wt1 = WT_ih1;
        const __bf16* wt2 = WT_hh1;
        const float*  gb  = gbase_in;
        const float*  w0r = w0row_in;
        const float*  bi1 = b_ih1_in;
        const float*  bh1 = b_hh1_in;
        const float*  wo1 = Wo1_in;
        const float*  pb1 = bo1_in;
        const float*  wo2 = Wo2_in;
        const float*  pb2 = bo2_in;
        OPAQUE_PTR(wt0); OPAQUE_PTR(wt1); OPAQUE_PTR(wt2);
        OPAQUE_PTR(gb);  OPAQUE_PTR(w0r);
        OPAQUE_PTR(bi1); OPAQUE_PTR(bh1);
        OPAQUE_PTR(wo1); OPAQUE_PTR(pb1); OPAQUE_PTR(wo2); OPAQUE_PTR(pb2);

        // ================= layer 0 GEMM: acc = h0 @ W_hh0   (tile ti = g*2+st)
        v8f acc[8];
        #pragma unroll
        for (int ti = 0; ti < 8; ++ti) acc[ti] = (v8f)0.0f;

        for (int k = 0; k < 16; ++k) {
            v16bf a = load_frag_lds(&h0s[ln * HID + k * 32 + ko8]);   // ds_load_b128 x2
            #pragma unroll
            for (int g = 0; g < 4; ++g)
                #pragma unroll
                for (int st = 0; st < 2; ++st) {
                    int col0 = g * HID + hbase + st * 16;
                    v16bf b = load_frag_g(&wt0[(size_t)(col0 + ln) * HID + k * 32 + ko8]);
                    acc[g * 2 + st] = WMMA_BF16(a, b, acc[g * 2 + st]);
                }
        }
        __syncthreads();   // everyone done reading old h0s

        // activations layer 0: fold in gates0_base + intent*w0row here (short live ranges)
        #pragma unroll
        for (int st = 0; st < 2; ++st) {
            int hcol = hbase + st * 16 + ln;
            #pragma unroll
            for (int r = 0; r < 8; ++r) {
                int m = r + 8 * lhalf;
                const float* gbr = gb + (size_t)(row0 + m) * GATES;
                float itv = intent_s[m];
                float iv = fsig (acc[0 + st][r] + ldg_f(gbr + 0 * HID + hcol) + itv * ldg_f(w0r + 0 * HID + hcol));
                float fv = fsig (acc[2 + st][r] + ldg_f(gbr + 1 * HID + hcol) + itv * ldg_f(w0r + 1 * HID + hcol));
                float gv = ftanh(acc[4 + st][r] + ldg_f(gbr + 2 * HID + hcol) + itv * ldg_f(w0r + 2 * HID + hcol));
                float ov = fsig (acc[6 + st][r] + ldg_f(gbr + 3 * HID + hcol) + itv * ldg_f(w0r + 3 * HID + hcol));
                float c  = fv * c0r[st][r] + iv * gv;
                c0r[st][r] = c;
                h0s[m * HID + hcol] = (__bf16)(ov * ftanh(c));
            }
        }
        __syncthreads();   // new h0s visible to all waves

        // ================= layer 1 GEMM: acc = h0n @ W_ih1 + h1 @ W_hh1
        #pragma unroll
        for (int ti = 0; ti < 8; ++ti) acc[ti] = (v8f)0.0f;

        for (int k = 0; k < 16; ++k) {
            v16bf a0 = load_frag_lds(&h0s[ln * HID + k * 32 + ko8]);
            v16bf a1 = load_frag_lds(&h1s[ln * HID + k * 32 + ko8]);
            #pragma unroll
            for (int g = 0; g < 4; ++g)
                #pragma unroll
                for (int st = 0; st < 2; ++st) {
                    int ti   = g * 2 + st;
                    int col0 = g * HID + hbase + st * 16;
                    v16bf b0 = load_frag_g(&wt1[(size_t)(col0 + ln) * HID + k * 32 + ko8]);
                    acc[ti] = WMMA_BF16(a0, b0, acc[ti]);
                    v16bf b1 = load_frag_g(&wt2[(size_t)(col0 + ln) * HID + k * 32 + ko8]);
                    acc[ti] = WMMA_BF16(a1, b1, acc[ti]);
                }
        }
        __syncthreads();

        // activations layer 1: fold in biases here
        #pragma unroll
        for (int st = 0; st < 2; ++st) {
            int hcol = hbase + st * 16 + ln;
            #pragma unroll
            for (int r = 0; r < 8; ++r) {
                int m = r + 8 * lhalf;
                float iv = fsig (acc[0 + st][r] + ldg_f(bi1 + 0 * HID + hcol) + ldg_f(bh1 + 0 * HID + hcol));
                float fv = fsig (acc[2 + st][r] + ldg_f(bi1 + 1 * HID + hcol) + ldg_f(bh1 + 1 * HID + hcol));
                float gv = ftanh(acc[4 + st][r] + ldg_f(bi1 + 2 * HID + hcol) + ldg_f(bh1 + 2 * HID + hcol));
                float ov = fsig (acc[6 + st][r] + ldg_f(bi1 + 3 * HID + hcol) + ldg_f(bh1 + 3 * HID + hcol));
                float c  = fv * c1r[st][r] + iv * gv;
                c1r[st][r] = c;
                h1s[m * HID + hcol] = (__bf16)(ov * ftanh(c));
            }
        }
        __syncthreads();   // new h1s visible for the head

        // ================= step head: intent = sigmoid(relu(h1n@Wo1+bo1)@Wo2+bo2)
        // wave w handles row m = w; lane j computes hidden unit j (32 units)
        {
            int m = wave;
            float hid = ldg_f(pb1 + lane);
            for (int k2 = 0; k2 < HID; ++k2)
                hid += (float)h1s[m * HID + k2] * ldg_f(wo1 + k2 * 32 + lane);
            hid = fmaxf(hid, 0.0f);
            float part = hid * ldg_f(wo2 + lane);
            #pragma unroll
            for (int off = 16; off; off >>= 1) part += __shfl_xor(part, off, 32);
            if (lane == 0) {
                float it = fsig(part + ldg_f(pb2));
                intent_s[m] = it;
                out_steps[(size_t)(row0 + m) * TSTEPS + t] = it;
            }
        }
        __syncthreads();   // intent_s ready for next step
    }

    // ================= global head on final h1
    {
        int m = wave;
        float hid = bg1[lane];
        for (int k2 = 0; k2 < HID; ++k2)
            hid += (float)h1s[m * HID + k2] * Wg1[k2 * 32 + lane];
        hid = fmaxf(hid, 0.0f);
        float part = hid * Wg2[lane];
        #pragma unroll
        for (int off = 16; off; off >>= 1) part += __shfl_xor(part, off, 32);
        if (lane == 0) out_global[row0 + m] = fsig(part + bg2[0]);
    }
}

// ---------------------------------------------------------------------------
extern "C" void kernel_launch(void* const* d_in, const int* in_sizes, int n_in,
                              void* d_out, int out_size, void* d_ws, size_t ws_size,
                              hipStream_t stream) {
    const float* enc         = (const float*)d_in[0];
    const float* ctx         = (const float*)d_in[1];
    const float* init_intent = (const float*)d_in[2];
    const float* Wh_init     = (const float*)d_in[3];
    const float* bh_init     = (const float*)d_in[4];
    const float* Wc_init     = (const float*)d_in[5];
    const float* bc_init     = (const float*)d_in[6];
    const float* W_ih0       = (const float*)d_in[7];
    const float* W_hh0       = (const float*)d_in[8];
    const float* b_ih0       = (const float*)d_in[9];
    const float* b_hh0       = (const float*)d_in[10];
    const float* W_ih1       = (const float*)d_in[11];
    const float* W_hh1       = (const float*)d_in[12];
    const float* b_ih1       = (const float*)d_in[13];
    const float* b_hh1       = (const float*)d_in[14];
    const float* Wo1         = (const float*)d_in[15];
    const float* bo1         = (const float*)d_in[16];
    const float* Wo2         = (const float*)d_in[17];
    const float* bo2         = (const float*)d_in[18];
    const float* Wg1         = (const float*)d_in[19];
    const float* bg1         = (const float*)d_in[20];
    const float* Wg2         = (const float*)d_in[21];
    const float* bg2         = (const float*)d_in[22];

    char* ws = (char*)d_ws;
    __bf16* WT    = (__bf16*)(ws + OFF_WT0);
    __bf16* WT0   = WT;
    __bf16* WT1   = (__bf16*)(ws + OFF_WT1);
    __bf16* WT2   = (__bf16*)(ws + OFF_WT2);
    float*  gbase = (float*)(ws + OFF_GB);
    float*  H0    = (float*)(ws + OFF_H0);
    float*  C0    = (float*)(ws + OFF_C0);
    float*  H1    = (float*)(ws + OFF_H1);
    float*  C1    = (float*)(ws + OFF_C1);

    float* out_steps  = (float*)d_out;
    float* out_global = out_steps + (size_t)BATCH * TSTEPS;

    k_prep_weights<<<dim3((GATES * HID) / 256, 3), 256, 0, stream>>>(W_hh0, W_ih1, W_hh1, WT);
    k_init_state<<<(BATCH * 1024) / 256, 256, 0, stream>>>(enc, Wh_init, bh_init, Wc_init, bc_init,
                                                           H0, C0, H1, C1);
    k_init_gbase<<<(BATCH * GATES) / 256, 256, 0, stream>>>(ctx, W_ih0, b_ih0, b_hh0, gbase);

    lstm_persistent<<<BATCH / 16, 512, 0, stream>>>(
        gbase, W_ih0 /* row 0 */, WT0, WT1, WT2,
        b_ih1, b_hh1, H0, C0, H1, C1, init_intent,
        Wo1, bo1, Wo2, bo2, Wg1, bg1, Wg2, bg2,
        out_steps, out_global);
}
